// SelfAttentionLayer_47124381172316
// MI455X (gfx1250) — matrix operand order repaired and checked by reading, not compiled, not run on hardware
//
#include <hip/hip_runtime.h>
#include <stdint.h>

typedef __attribute__((ext_vector_type(16))) __bf16 v16bf;
typedef __attribute__((ext_vector_type(8)))  float  v8f;
typedef __attribute__((ext_vector_type(4)))  int    i32x4;

#define B_   8
#define S_   4096
#define D_   128
#define BLKQ 128
#define BLKK 32

#define AS1 __attribute__((address_space(1)))
#define AS3 __attribute__((address_space(3)))

#if defined(__AMDGCN__) && __has_builtin(__builtin_amdgcn_global_load_async_to_lds_b128)
#define ASYNC_COPY 1
#endif

static __device__ __forceinline__ unsigned short f2bf(float f) {
    union { float f; unsigned u; } x; x.f = f;
    unsigned u = x.u;
    return (unsigned short)((u + 0x7FFFu + ((u >> 16) & 1u)) >> 16);
}

// pack two f32 -> packed bf16x2 (hardware cvt_pk if the builtin exists)
static __device__ __forceinline__ unsigned pack2bf(float lo, float hi) {
#if defined(__AMDGCN__) && __has_builtin(__builtin_amdgcn_cvt_pk_bf16_f32)
    auto r = __builtin_amdgcn_cvt_pk_bf16_f32(lo, hi);
    union { decltype(r) v; unsigned u; } x; x.v = r;
    return x.u;
#else
    return (unsigned)f2bf(lo) | ((unsigned)f2bf(hi) << 16);
#endif
}

static __device__ __forceinline__ void wait_async0() {
#ifdef ASYNC_COPY
#if __has_builtin(__builtin_amdgcn_s_wait_asynccnt)
    __builtin_amdgcn_s_wait_asynccnt(0);
#else
    asm volatile("s_wait_asynccnt 0x0" ::: "memory");
#endif
#endif
}

union Frag16 { uint4 q[2]; v16bf v; };

static __device__ __forceinline__ v8f wmma_bf16(v16bf a, v16bf b, v8f c) {
    return __builtin_amdgcn_wmma_f32_16x16x32_bf16(
        /*neg_a=*/false, a, /*neg_b=*/false, b,
        /*c_mod=*/(short)0, c, /*reuse_a=*/false, /*reuse_b=*/false);
}

// ---------------------------------------------------------------------------
// Kernel 1: Q/K/V projection.  One block = 128 rows of one batch.
// ---------------------------------------------------------------------------
__global__ __launch_bounds__(256) void qkv_proj_kernel(
    const float* __restrict__ X, const float* __restrict__ W,
    unsigned short* __restrict__ Q, unsigned short* __restrict__ K,
    unsigned short* __restrict__ V)
{
    __shared__ __align__(16) unsigned short Xs[128 * 128];
    __shared__ __align__(16) unsigned short Wt[128 * 128]; // transposed: [dout][din]

    const int t    = threadIdx.x;
    const int wid  = t >> 5;
    const int lane = t & 31;
    const int n    = lane & 15;
    const int g    = lane >> 4;
    const int b    = blockIdx.y;
    const int row0 = blockIdx.x * BLKQ;

    const float* Xblk = X + ((size_t)b * S_ + row0) * D_;
    {   // packed f32x2 -> bf16x2 conversion into LDS
        const float2* xs = reinterpret_cast<const float2*>(Xblk);
        unsigned* xd = reinterpret_cast<unsigned*>(Xs);
        for (int i = t; i < 128 * 128 / 2; i += 256) {
            float2 xv = xs[i];
            xd[i] = pack2bf(xv.x, xv.y);
        }
    }
    __syncthreads();

    // A fragments for this wave's 16 rows (16x32 bf16 each)
    Frag16 a[4];
    {
        const int m = wid * 16 + n;
#pragma unroll
        for (int kk = 0; kk < 4; ++kk) {
            a[kk].q[0] = *reinterpret_cast<const uint4*>(&Xs[m * 128 + 32 * kk +      8 * g]);
            a[kk].q[1] = *reinterpret_cast<const uint4*>(&Xs[m * 128 + 32 * kk + 16 + 8 * g]);
        }
    }

    for (int w = 0; w < 3; ++w) {
        __syncthreads();
        const float* Wm = W + (size_t)w * D_ * D_;
        {   // transpose + convert, paired along din so LDS stores are 32-bit
            unsigned* wd = reinterpret_cast<unsigned*>(Wt);
            for (int i = t; i < 128 * 128 / 2; i += 256) {
                const int din = (i & 63) * 2, dout = i >> 6;
                wd[dout * 64 + (din >> 1)] =
                    pack2bf(Wm[din * 128 + dout], Wm[(din + 1) * 128 + dout]);
            }
        }
        __syncthreads();

        unsigned short* Out = (w == 0) ? Q : (w == 1) ? K : V;
#pragma unroll
        for (int j = 0; j < 8; ++j) {
            v8f c = {};
#pragma unroll
            for (int kk = 0; kk < 4; ++kk) {
                Frag16 bf;
                const int base = (16 * j + n) * 128 + 32 * kk + 8 * g;
                bf.q[0] = *reinterpret_cast<const uint4*>(&Wt[base]);
                bf.q[1] = *reinterpret_cast<const uint4*>(&Wt[base + 16]);
                c = wmma_bf16(a[kk].v, bf.v, c);
            }
#pragma unroll
            for (int v = 0; v < 8; ++v) {
                const int m = wid * 16 + v + 8 * g;
                Out[((size_t)b * S_ + row0 + m) * D_ + 16 * j + n] = f2bf(c[v]);
            }
        }
    }
}

// ---------------------------------------------------------------------------
// Kernel 2: flash attention + per-block column sums for the mean pool.
// ---------------------------------------------------------------------------
__global__ __launch_bounds__(256) void attn_kernel(
    const unsigned short* __restrict__ Q, const unsigned short* __restrict__ K,
    const unsigned short* __restrict__ V, float* __restrict__ partial)
{
    __shared__ __align__(16) unsigned short Ks[BLKK * 128];     // [key][d]
    __shared__ __align__(16) unsigned short Vs[128 * BLKK];     // transposed [d][key]
    __shared__ __align__(16) unsigned short Ps[8][16 * 32];     // per-wave P strip

    const int t    = threadIdx.x;
    const int wid  = t >> 5;
    const int lane = t & 31;
    const int n    = lane & 15;
    const int g    = lane >> 4;
    const int b    = blockIdx.y;
    const int q0   = blockIdx.x * BLKQ;

    // Q A-fragments
    Frag16 qa[4];
    {
        const int qrow = q0 + wid * 16 + n;
        const unsigned short* qp = Q + ((size_t)b * S_ + qrow) * D_;
#pragma unroll
        for (int kk = 0; kk < 4; ++kk) {
            qa[kk].q[0] = *reinterpret_cast<const uint4*>(qp + 32 * kk +      8 * g);
            qa[kk].q[1] = *reinterpret_cast<const uint4*>(qp + 32 * kk + 16 + 8 * g);
        }
    }

    v8f   acc[8];
    float mrow[8], lrow[8];
#pragma unroll
    for (int j = 0; j < 8; ++j) acc[j] = {};
#pragma unroll
    for (int v = 0; v < 8; ++v) { mrow[v] = -3.0e38f; lrow[v] = 0.0f; }

    for (int kb = 0; kb < S_ / BLKK; ++kb) {
        __syncthreads();
        {   // stage K block into LDS (async DMA if available), V block transposed
            const unsigned short* ksrc = K + ((size_t)b * S_ + kb * BLKK) * D_;
#ifdef ASYNC_COPY
            for (int i = t; i < BLKK * 128 / 8; i += 256) {
                unsigned short* gsrc = const_cast<unsigned short*>(ksrc + i * 8);
                __builtin_amdgcn_global_load_async_to_lds_b128(
                    (AS1 i32x4*)(void*)gsrc,
                    (AS3 i32x4*)(void*)(&Ks[i * 8]), 0, 0);
            }
#else
            {
                const uint4* src = reinterpret_cast<const uint4*>(ksrc);
                uint4* dst = reinterpret_cast<uint4*>(Ks);
                for (int i = t; i < BLKK * 128 / 8; i += 256) dst[i] = src[i];
            }
#endif
            const unsigned short* vsrc = V + ((size_t)b * S_ + kb * BLKK) * D_;
            unsigned* vd = reinterpret_cast<unsigned*>(Vs);
            for (int i = t; i < BLKK * 128 / 2; i += 256) {
                const int key = (i & 15) * 2, d = i >> 4;
                // pack two consecutive keys for column d
                vd[d * 16 + (key >> 1)] =
                    (unsigned)vsrc[key * 128 + d] |
                    ((unsigned)vsrc[(key + 1) * 128 + d] << 16);
            }
            // prefetch next block's K/V lines while this block computes
            if (kb + 1 < S_ / BLKK) {
                const unsigned short* kn = K + ((size_t)b * S_ + (kb + 1) * BLKK) * D_;
                const unsigned short* vn = V + ((size_t)b * S_ + (kb + 1) * BLKK) * D_;
                __builtin_prefetch(kn + t * 16, 0, 0);
                __builtin_prefetch(vn + t * 16, 0, 0);
            }
            wait_async0();
        }
        __syncthreads();

        // S = Q * K^T  (two 16x16 tiles covering 32 keys)
        v8f s0 = {}, s1 = {};
#pragma unroll
        for (int kk = 0; kk < 4; ++kk) {
            Frag16 b0, b1;
            const int base0 = n * 128 + 32 * kk + 8 * g;
            const int base1 = (16 + n) * 128 + 32 * kk + 8 * g;
            b0.q[0] = *reinterpret_cast<const uint4*>(&Ks[base0]);
            b0.q[1] = *reinterpret_cast<const uint4*>(&Ks[base0 + 16]);
            b1.q[0] = *reinterpret_cast<const uint4*>(&Ks[base1]);
            b1.q[1] = *reinterpret_cast<const uint4*>(&Ks[base1 + 16]);
            s0 = wmma_bf16(qa[kk].v, b0.v, s0);
            s1 = wmma_bf16(qa[kk].v, b1.v, s1);
        }

        // online softmax (row = v + 8g; 16-lane half-wave reductions)
        float p0[8], p1[8], alpha[8];
#pragma unroll
        for (int v = 0; v < 8; ++v) {
            float r = fmaxf(s0[v], s1[v]);
            r = fmaxf(r, __shfl_xor(r, 8));
            r = fmaxf(r, __shfl_xor(r, 4));
            r = fmaxf(r, __shfl_xor(r, 2));
            r = fmaxf(r, __shfl_xor(r, 1));
            const float mnew = fmaxf(mrow[v], r);
            alpha[v] = __expf(mrow[v] - mnew);
            p0[v] = __expf(s0[v] - mnew);
            p1[v] = __expf(s1[v] - mnew);
            float rs = p0[v] + p1[v];
            rs += __shfl_xor(rs, 8);
            rs += __shfl_xor(rs, 4);
            rs += __shfl_xor(rs, 2);
            rs += __shfl_xor(rs, 1);
            lrow[v] = lrow[v] * alpha[v] + rs;
            mrow[v] = mnew;
        }
#pragma unroll
        for (int j = 0; j < 8; ++j)
#pragma unroll
            for (int v = 0; v < 8; ++v) acc[j][v] *= alpha[v];

        // C-layout -> A-layout re-stripe of P via per-wave LDS strip.
        // Same-wave DS ops are executed in order by the LDS unit, so only a
        // compiler reordering fence is needed (no block barrier).
#pragma unroll
        for (int v = 0; v < 8; ++v) {
            const int row = v + 8 * g;
            Ps[wid][row * 32 + n]      = f2bf(p0[v]);
            Ps[wid][row * 32 + 16 + n] = f2bf(p1[v]);
        }
        asm volatile("" ::: "memory");

        Frag16 pa;
        pa.q[0] = *reinterpret_cast<const uint4*>(&Ps[wid][n * 32 +      8 * g]);
        pa.q[1] = *reinterpret_cast<const uint4*>(&Ps[wid][n * 32 + 16 + 8 * g]);

        // acc += P * V
#pragma unroll
        for (int j = 0; j < 8; ++j) {
            Frag16 bv;
            const int base = (16 * j + n) * 32 + 8 * g;
            bv.q[0] = *reinterpret_cast<const uint4*>(&Vs[base]);
            bv.q[1] = *reinterpret_cast<const uint4*>(&Vs[base + 16]);
            acc[j] = wmma_bf16(pa.v, bv.v, acc[j]);
        }
    }

    // finalize: divide by row sums, column-sum this wave's 16 rows, store partials
    float rcp[8];
#pragma unroll
    for (int v = 0; v < 8; ++v) rcp[v] = 1.0f / lrow[v];
#pragma unroll
    for (int j = 0; j < 8; ++j) {
        float csum = 0.0f;
#pragma unroll
        for (int v = 0; v < 8; ++v) csum += acc[j][v] * rcp[v];
        csum += __shfl_xor(csum, 16);     // combine the two half-wave row groups
        if (lane < 16) {
            const size_t slot = ((size_t)(b * 32 + blockIdx.x) * 8 + wid) * 128;
            partial[slot + 16 * j + lane] = csum;
        }
    }
}

// ---------------------------------------------------------------------------
// Kernel 3: deterministic mean-pool reduction (fixed summation order).
// ---------------------------------------------------------------------------
__global__ void reduce_kernel(const float* __restrict__ partial, float* __restrict__ out) {
    const int i = blockIdx.x * blockDim.x + threadIdx.x; // 0..1023
    if (i >= B_ * D_) return;
    const int b = i >> 7, d = i & 127;
    float s = 0.0f;
    for (int q = 0; q < 32 * 8; ++q)
        s += partial[((size_t)(b * 32 * 8 + q)) * 128 + d];
    out[i] = s * (1.0f / (float)S_);
}

// ---------------------------------------------------------------------------
extern "C" void kernel_launch(void* const* d_in, const int* in_sizes, int n_in,
                              void* d_out, int out_size, void* d_ws, size_t ws_size,
                              hipStream_t stream) {
    (void)in_sizes; (void)n_in; (void)ws_size;
    const float* X = (const float*)d_in[0];   // [8,4096,128] fp32
    const float* W = (const float*)d_in[1];   // [3,128,128]  fp32
    float* out = (float*)d_out;               // [8,128]      fp32

    const size_t NE = (size_t)B_ * S_ * D_;   // 4,194,304 elements
    unsigned short* Qb = (unsigned short*)d_ws;
    unsigned short* Kb = Qb + NE;
    unsigned short* Vb = Kb + NE;
    float* Pp = (float*)(Vb + NE);            // partial sums: 8*32*8*128 floats

    dim3 grid(S_ / BLKQ, B_), block(256);
    qkv_proj_kernel<<<grid, block, 0, stream>>>(X, W, Qb, Kb, Vb);
    attn_kernel<<<grid, block, 0, stream>>>(Qb, Kb, Vb, Pp);
    reduce_kernel<<<dim3((out_size + 255) / 256), dim3(256), 0, stream>>>(Pp, out);
}
